// MashDecoderV2_77137612636653
// MI455X (gfx1250) — matrix operand order repaired and checked by reading, not compiled, not run on hardware
//
#include <hip/hip_runtime.h>

// MashDecoderV2 for gfx1250 (MI455X).
// K0: single-workgroup 256-layer Mamba stack (latency-bound, LDS-resident x).
//     Also emits K and V *pre-swizzled into CDNA5 WMMA B-fragment layout*
//     (f16, zero-padded) so the attention kernel loads operands with b128s.
// K1: per-wave 16-query tile cross-attention with v_wmma_f32_16x16x32_f16
//     (Q.K^T and P.V), softmax in LDS, then VALU tail (ca_out/LN/GEGLU-FF/occ).

#define D_MODEL 40
#define N_LAYER 256
#define D_INNER 80
#define D_STATE 16
#define D_CONV  4
#define DT_RANK 3
#define LSEQ    400
#define NQ      65536
#define SCPAD   416   // 400 keys padded to multiple of 32 for P.V WMMA K-dim
#define NKT     25    // key tiles of 16
#define NKC     13    // K-chunks of 32 for P.V

typedef __attribute__((ext_vector_type(16))) _Float16 v16h;
typedef __attribute__((ext_vector_type(8)))  _Float16 v8h;
typedef __attribute__((ext_vector_type(8)))  float    v8f;

union H16 { v16h v; v8h g[2]; _Float16 h[16]; };
union F8  { v8f  v; float f[8]; };

// fragment element counts
#define KFRAG_N (NKT * 2 * 32 * 16)   // 25600 halfs
#define VFRAG_N (NKC * 3 * 32 * 16)   // 19968 halfs

__device__ __forceinline__ float sigm(float x) { return 1.0f / (1.0f + __expf(-x)); }

// ---------------------------------------------------------------------------
// K0: Mamba stack (one block, 1024 threads), plus context LN + KV projection
//     emitted directly into WMMA B-fragment layout.
// ---------------------------------------------------------------------------
__global__ __launch_bounds__(1024)
void mamba_stack_kernel(const float* __restrict__ x0,
                        const float* __restrict__ l_norm_w,
                        const float* __restrict__ l_in_w,
                        const float* __restrict__ l_conv_w,
                        const float* __restrict__ l_conv_b,
                        const float* __restrict__ l_xproj_w,
                        const float* __restrict__ l_dt_w,
                        const float* __restrict__ l_dt_b,
                        const float* __restrict__ l_A_log,
                        const float* __restrict__ l_D,
                        const float* __restrict__ l_out_w,
                        const float* __restrict__ ca_lnc_w,
                        const float* __restrict__ ca_lnc_b,
                        const float* __restrict__ ca_kv_w,
                        float* __restrict__ xin,    // 400*80 (pre-conv x, later y)
                        float* __restrict__ zbuf,   // 400*80
                        float* __restrict__ xsb,    // 400*80 (silu(conv(x)))
                        float* __restrict__ dbc,    // 400*35
                        float* __restrict__ rmsb,   // 400
                        _Float16* __restrict__ kfrag,  // [25][2][32][16]
                        _Float16* __restrict__ vfrag)  // [13][3][32][16]
{
  __shared__ float xbuf[LSEQ * D_MODEL];   // 64000 B: residual stream
  const int tid = threadIdx.x;

  for (int i = tid; i < LSEQ * D_MODEL; i += 1024) xbuf[i]  = x0[i];
  for (int i = tid; i < KFRAG_N;        i += 1024) kfrag[i] = (_Float16)0.0f;
  for (int i = tid; i < VFRAG_N;        i += 1024) vfrag[i] = (_Float16)0.0f;
  __syncthreads();

  for (int layer = 0; layer < N_LAYER; ++layer) {
    const float* norm_w  = l_norm_w  + layer * D_MODEL;
    const float* in_w    = l_in_w    + layer * 2 * D_INNER * D_MODEL;
    const float* conv_w  = l_conv_w  + layer * D_INNER * D_CONV;
    const float* conv_b  = l_conv_b  + layer * D_INNER;
    const float* xproj_w = l_xproj_w + layer * (DT_RANK + 2 * D_STATE) * D_INNER;
    const float* dt_w    = l_dt_w    + layer * D_INNER * DT_RANK;
    const float* dt_b    = l_dt_b    + layer * D_INNER;
    const float* A_log   = l_A_log   + layer * D_INNER * D_STATE;
    const float* Dp      = l_D       + layer * D_INNER;
    const float* out_w   = l_out_w   + layer * D_MODEL * D_INNER;

    // phase 1: rms factor per position
    for (int l = tid; l < LSEQ; l += 1024) {
      float ss = 0.0f;
      for (int d = 0; d < D_MODEL; ++d) { float v = xbuf[l * D_MODEL + d]; ss += v * v; }
      rmsb[l] = rsqrtf(ss * (1.0f / D_MODEL) + 1e-5f);
    }
    __syncthreads();

    // phase 2: in-projection xz = rmsnorm(x) @ in_w^T  (split into x / z)
    for (int idx = tid; idx < LSEQ * 2 * D_INNER; idx += 1024) {
      int l = idx / (2 * D_INNER), e = idx % (2 * D_INNER);
      const float* wrow = in_w + e * D_MODEL;
      float acc = 0.0f;
      for (int d = 0; d < D_MODEL; ++d) acc += xbuf[l * D_MODEL + d] * norm_w[d] * wrow[d];
      acc *= rmsb[l];
      if (e < D_INNER) xin[l * D_INNER + e] = acc;
      else             zbuf[l * D_INNER + (e - D_INNER)] = acc;
    }
    __syncthreads();

    // phase 3: causal depthwise conv (k=4) + SiLU
    for (int idx = tid; idx < LSEQ * D_INNER; idx += 1024) {
      int l = idx / D_INNER, c = idx % D_INNER;
      float acc = conv_b[c];
      for (int j = 0; j < D_CONV; ++j) {
        int ls = l - (D_CONV - 1) + j;
        if (ls >= 0) acc += xin[ls * D_INNER + c] * conv_w[c * D_CONV + j];
      }
      xsb[idx] = acc * sigm(acc);
    }
    __syncthreads();

    // phase 4: x-projection -> [dtr(3) | B(16) | C(16)]
    for (int idx = tid; idx < LSEQ * 35; idx += 1024) {
      int l = idx / 35, e = idx % 35;
      const float* wr = xproj_w + e * D_INNER;
      float acc = 0.0f;
      for (int d = 0; d < D_INNER; ++d) acc += xsb[l * D_INNER + d] * wr[d];
      dbc[idx] = acc;
    }
    __syncthreads();

    // phase 5: selective scan (80 channel threads, 16 states each, seq over L)
    if (tid < D_INNER) {
      const int d = tid;
      float A[D_STATE], h[D_STATE];
      for (int s = 0; s < D_STATE; ++s) { A[s] = -__expf(A_log[d * D_STATE + s]); h[s] = 0.0f; }
      const float w0 = dt_w[d * 3 + 0], w1 = dt_w[d * 3 + 1], w2 = dt_w[d * 3 + 2];
      const float bb = dt_b[d], Dd = Dp[d];
      for (int l = 0; l < LSEQ; ++l) {
        const float* row = dbc + l * 35;
        float pre = row[0] * w0 + row[1] * w1 + row[2] * w2 + bb;
        float dt  = (pre > 20.0f) ? pre : log1pf(__expf(pre));   // softplus
        float xv  = xsb[l * D_INNER + d];
        float y   = xv * Dd;
        for (int s = 0; s < D_STATE; ++s) {
          h[s] = __expf(dt * A[s]) * h[s] + dt * row[3 + s] * xv;
          y += h[s] * row[19 + s];
        }
        float zv = zbuf[l * D_INNER + d];
        xin[l * D_INNER + d] = y * zv * sigm(zv);   // y * silu(z), reuse xin
      }
    }
    __syncthreads();

    // phase 6: out-projection + residual
    for (int idx = tid; idx < LSEQ * D_MODEL; idx += 1024) {
      int l = idx / D_MODEL, e = idx % D_MODEL;
      const float* wr = out_w + e * D_INNER;
      float acc = 0.0f;
      for (int d = 0; d < D_INNER; ++d) acc += xin[l * D_INNER + d] * wr[d];
      xbuf[idx] += acc;
    }
    __syncthreads();
  }

  // Context layernorm + KV projection; scatter into WMMA B-fragment layouts.
  // B layout (16-bit, 32x16): lanes 0-15 hold K in {0..7,16..23}, lanes 16-31 K in {8..15,24..31},
  // N = lane%16, element = (K&7) | ((K>>4)<<3).
  for (int idx = tid; idx < LSEQ * 2 * D_MODEL; idx += 1024) {
    int l = idx / (2 * D_MODEL), e = idx % (2 * D_MODEL);
    float m = 0.0f;
    for (int d = 0; d < D_MODEL; ++d) m += xbuf[l * D_MODEL + d];
    m *= (1.0f / D_MODEL);
    float v = 0.0f;
    for (int d = 0; d < D_MODEL; ++d) { float t = xbuf[l * D_MODEL + d] - m; v += t * t; }
    float inv = rsqrtf(v * (1.0f / D_MODEL) + 1e-5f);
    const float* wr = ca_kv_w + e * D_MODEL;
    float acc = 0.0f;
    for (int d = 0; d < D_MODEL; ++d)
      acc += ((xbuf[l * D_MODEL + d] - m) * inv * ca_lnc_w[d] + ca_lnc_b[d]) * wr[d];
    if (e < D_MODEL) {
      // K fragment slot: B[k=e (dim), n=l (key)] within tile kt=l/16, chunk e/32
      int kt = l >> 4, nn = l & 15;
      int ch = e >> 5, dd = e & 31;
      int lane_ = (((dd >> 3) & 1) ? 16 : 0) + nn;
      kfrag[(((kt * 2 + ch) * 32) + lane_) * 16 + ((dd & 7) | ((dd >> 4) << 3))] = (_Float16)acc;
    } else {
      // V fragment slot: B[k=l (key), n=e-40 (dim)] within chunk kc=l/32, ntile
      int d2 = e - D_MODEL;
      int kc = l >> 5, rr = l & 31;
      int t  = d2 >> 4, cc = d2 & 15;
      int lane_ = (((rr >> 3) & 1) ? 16 : 0) + cc;
      vfrag[(((kc * 3 + t) * 32) + lane_) * 16 + ((rr & 7) | ((rr >> 4) << 3))] = (_Float16)acc;
    }
  }
}

// ---------------------------------------------------------------------------
// K1: WMMA cross-attention + FF. 64 threads (2 waves), 16 queries per wave.
// ---------------------------------------------------------------------------
__global__ __launch_bounds__(64)
void attn_ff_kernel(const float* __restrict__ qry,
                    const float* __restrict__ pe_mlp_w, const float* __restrict__ pe_mlp_b,
                    const float* __restrict__ lnq_w,    const float* __restrict__ lnq_b,
                    const float* __restrict__ ca_q_w,
                    const float* __restrict__ ca_out_w, const float* __restrict__ ca_out_b,
                    const float* __restrict__ ff_ln_w,  const float* __restrict__ ff_ln_b,
                    const float* __restrict__ fc1_w,    const float* __restrict__ fc1_b,
                    const float* __restrict__ fc2_w,    const float* __restrict__ fc2_b,
                    const float* __restrict__ out_w,    const float* __restrict__ out_b,
                    const _Float16* __restrict__ kfrag, // [25][2][32][16] f16
                    const _Float16* __restrict__ vfrag, // [13][3][32][16] f16
                    float* __restrict__ out)
{
  // Q tile stored directly in WMMA A-fragment layout: [wave][chunk][lane][16]
  __shared__ __attribute__((aligned(32))) _Float16 qfrag[2][2][32][16];  //  4096 B
  __shared__ __attribute__((aligned(32))) float    sc[2][16][SCPAD];     // 53248 B
  __shared__ __attribute__((aligned(32))) float    latb[2][16][48];      //  6144 B

  const int tid    = threadIdx.x;
  const int wv     = tid >> 5;
  const int lane   = tid & 31;
  const int ln16   = lane & 15;
  const bool lohalf = (lane < 16);

  __builtin_prefetch(kfrag, 0, 3);   // global_prefetch_b8: K/V fragments L2-hot
  __builtin_prefetch(vfrag, 0, 3);

  // ---- Step A: PE-MLP + LNq + Q-projection; scatter into A-frag layout ---
  // A layout (16-bit 16x32): lanes 0-15 M=0..15 K in {0..7,16..23};
  //                          lanes 16-31 same M, K in {8..15,24..31}.
  if (lane < 16) {
    const int n = blockIdx.x * 32 + wv * 16 + lane;
    const int m = lane;
    float qc[3] = { qry[n * 3 + 0], qry[n * 3 + 1], qry[n * 3 + 2] };
    float pe[51];
    const float PI = 3.14159265358979323846f;
    for (int a = 0; a < 3; ++a) {
      float f = PI;                      // 2^j * pi frequencies
      for (int j = 0; j < 8; ++j) {
        float p = qc[a] * f;
        pe[a * 8 + j]      = __sinf(p);
        pe[24 + a * 8 + j] = __cosf(p);
        f *= 2.0f;
      }
    }
    pe[48] = qc[0]; pe[49] = qc[1]; pe[50] = qc[2];
    float qe[40];
    for (int d = 0; d < 40; ++d) {
      const float* wr = pe_mlp_w + d * 51;
      float acc = pe_mlp_b[d];
      for (int f = 0; f < 51; ++f) acc += pe[f] * wr[f];
      qe[d] = acc;
    }
    float mm = 0.0f; for (int d = 0; d < 40; ++d) mm += qe[d]; mm *= (1.0f / 40.0f);
    float vv = 0.0f; for (int d = 0; d < 40; ++d) { float t = qe[d] - mm; vv += t * t; }
    float inv = rsqrtf(vv * (1.0f / 40.0f) + 1e-5f);
    float qn[40];
    for (int d = 0; d < 40; ++d) qn[d] = (qe[d] - mm) * inv * lnq_w[d] + lnq_b[d];
    for (int e = 0; e < 64; ++e) {
      float acc = 0.0f;
      if (e < 40) {
        const float* wr = ca_q_w + e * 40;
        for (int d = 0; d < 40; ++d) acc += qn[d] * wr[d];
      }
      int ch = e >> 5, dd = e & 31;
      int lane_ = (((dd >> 3) & 1) == 0) ? m : (16 + m);
      int elem  = (dd & 7) | ((dd >> 4) << 3);
      qfrag[wv][ch][lane_][elem] = (_Float16)acc;
    }
  }
  __syncthreads();

  // ---- Step B: A-fragments of Q: one contiguous 32B LDS load each --------
  H16 a0, a1;
  a0.v = *(const v16h*)&qfrag[wv][0][lane][0];
  a1.v = *(const v16h*)&qfrag[wv][1][lane][0];

  // ---- Step C: scores = Q.K^T via WMMA (25 key tiles x 2 K-chunks) -------
  const float scale = 0.15811388300841897f;   // 40^-0.5
  {
    const v16h* kf = (const v16h*)kfrag;      // [(kt*2+ch)*32 + lane]
    for (int kt = 0; kt < NKT; ++kt) {
      H16 b0, b1;
      b0.v = kf[(kt * 2 + 0) * 32 + lane];
      b1.v = kf[(kt * 2 + 1) * 32 + lane];
      v8f c = {};
      c = __builtin_amdgcn_wmma_f32_16x16x32_f16(false, a0.v, false, b0.v, (short)0, c, false, false);
      c = __builtin_amdgcn_wmma_f32_16x16x32_f16(false, a1.v, false, b1.v, (short)0, c, false, false);
      F8 cf; cf.v = c;
      const int mbase = lohalf ? 0 : 8;
      const int n0 = kt * 16;
      for (int r = 0; r < 8; ++r) sc[wv][mbase + r][n0 + ln16] = cf.f[r] * scale;
    }
  }
  __syncthreads();

  // ---- Step D: softmax per row; write P back in place as f16 + zero pad --
  if (lane < 16) {
    float* row = &sc[wv][lane][0];
    float mx = -3.0e38f;
    for (int n2 = 0; n2 < 400; ++n2) mx = fmaxf(mx, row[n2]);
    float sum = 0.0f;
    for (int n2 = 0; n2 < 400; ++n2) sum += __expf(row[n2] - mx);
    float inv = 1.0f / sum;
    _Float16* p = (_Float16*)row;   // f16 slot n2 overwrites f32 slot n2/2 (already read)
    for (int n2 = 0; n2 < 400; ++n2) {
      float sv = row[n2];
      p[n2] = (_Float16)(__expf(sv - mx) * inv);
    }
    for (int n2 = 400; n2 < SCPAD; ++n2) p[n2] = (_Float16)0.0f;
  }
  __syncthreads();

  // ---- Step E: lat = P @ V via WMMA (13 K-chunks x 3 N-tiles) ------------
  F8 acc0, acc1, acc2;
  {
    v8f c0 = {}, c1 = {}, c2 = {};
    const int klo8 = lohalf ? 0 : 8;
    const _Float16* prow = (const _Float16*)&sc[wv][ln16][0];
    const v16h* vf = (const v16h*)vfrag;      // [(kc*3+t)*32 + lane]
    for (int kc = 0; kc < NKC; ++kc) {
      const int kbase = kc * 32;
      H16 pa;
      pa.g[0] = *(const v8h*)(prow + kbase + klo8);        // 8 contiguous halfs
      pa.g[1] = *(const v8h*)(prow + kbase + 16 + klo8);   // 8 contiguous halfs
      H16 b0, b1, b2;
      b0.v = vf[(kc * 3 + 0) * 32 + lane];
      b1.v = vf[(kc * 3 + 1) * 32 + lane];
      b2.v = vf[(kc * 3 + 2) * 32 + lane];
      c0 = __builtin_amdgcn_wmma_f32_16x16x32_f16(false, pa.v, false, b0.v, (short)0, c0, false, false);
      c1 = __builtin_amdgcn_wmma_f32_16x16x32_f16(false, pa.v, false, b1.v, (short)0, c1, false, false);
      c2 = __builtin_amdgcn_wmma_f32_16x16x32_f16(false, pa.v, false, b2.v, (short)0, c2, false, false);
    }
    acc0.v = c0; acc1.v = c1; acc2.v = c2;
  }
  {
    const int mbase = lohalf ? 0 : 8;
    for (int r = 0; r < 8; ++r) {
      latb[wv][mbase + r][ln16]      = acc0.f[r];
      latb[wv][mbase + r][16 + ln16] = acc1.f[r];
      latb[wv][mbase + r][32 + ln16] = acc2.f[r];
    }
  }
  __syncthreads();

  // ---- Step F: tail, 2 threads per query (ca_out, LN, fc1+GEGLU, fc2, occ)
  {
    float* ab = &sc[0][0][0];          // reuse score LDS as [32][320] fc1 acts
    const int qi = tid >> 1, hf = tid & 1;
    const int w2 = qi >> 4, m2 = qi & 15;
    float lat[40];
    for (int d = 0; d < 40; ++d) lat[d] = latb[w2][m2][d];
    float lat2[40];
    for (int e = 0; e < 40; ++e) {
      const float* wr = ca_out_w + e * 40;
      float acc = ca_out_b[e];
      for (int d = 0; d < 40; ++d) acc += lat[d] * wr[d];
      lat2[e] = acc;
    }
    float m = 0.0f; for (int d = 0; d < 40; ++d) m += lat2[d]; m *= (1.0f / 40.0f);
    float v = 0.0f; for (int d = 0; d < 40; ++d) { float t = lat2[d] - m; v += t * t; }
    float inv = rsqrtf(v * (1.0f / 40.0f) + 1e-5f);
    float h[40];
    for (int d = 0; d < 40; ++d) h[d] = (lat2[d] - m) * inv * ff_ln_w[d] + ff_ln_b[d];
    for (int e = hf * 160; e < hf * 160 + 160; ++e) {
      const float* wr = fc1_w + e * 40;
      float acc = fc1_b[e];
      for (int d = 0; d < 40; ++d) acc += h[d] * wr[d];
      ab[qi * 320 + e] = acc;
    }
    __syncthreads();
    float part = 0.0f;
    for (int d = hf * 20; d < hf * 20 + 20; ++d) {
      const float* wr = fc2_w + d * 160;
      float acc = fc2_b[d];
      for (int f = 0; f < 160; ++f) {
        float x1 = ab[qi * 320 + f];
        float g  = ab[qi * 320 + 160 + f];
        float gg = 0.5f * g * (1.0f + erff(g * 0.70710678118f));  // exact gelu
        acc += x1 * gg * wr[f];
      }
      part += (lat2[d] + acc) * out_w[d];
    }
    part += __shfl_xor(part, 1, 32);
    if (hf == 0) out[blockIdx.x * 32 + qi] = part + out_b[0];
  }
}

// ---------------------------------------------------------------------------
extern "C" void kernel_launch(void* const* d_in, const int* in_sizes, int n_in,
                              void* d_out, int out_size, void* d_ws, size_t ws_size,
                              hipStream_t stream) {
  (void)in_sizes; (void)n_in; (void)out_size; (void)ws_size;
  const float* mash     = (const float*)d_in[0];
  const float* qry      = (const float*)d_in[1];
  const float* l_norm_w = (const float*)d_in[2];
  const float* l_in_w   = (const float*)d_in[3];
  const float* l_conv_w = (const float*)d_in[4];
  const float* l_conv_b = (const float*)d_in[5];
  const float* l_xproj  = (const float*)d_in[6];
  const float* l_dt_w   = (const float*)d_in[7];
  const float* l_dt_b   = (const float*)d_in[8];
  const float* l_A_log  = (const float*)d_in[9];
  const float* l_D      = (const float*)d_in[10];
  const float* l_out_w  = (const float*)d_in[11];
  const float* pe_mlp_w = (const float*)d_in[12];
  const float* pe_mlp_b = (const float*)d_in[13];
  const float* ca_lnq_w = (const float*)d_in[14];
  const float* ca_lnq_b = (const float*)d_in[15];
  const float* ca_lnc_w = (const float*)d_in[16];
  const float* ca_lnc_b = (const float*)d_in[17];
  const float* ca_q_w   = (const float*)d_in[18];
  const float* ca_kv_w  = (const float*)d_in[19];
  const float* ca_out_w = (const float*)d_in[20];
  const float* ca_out_b = (const float*)d_in[21];
  const float* ff_ln_w  = (const float*)d_in[22];
  const float* ff_ln_b  = (const float*)d_in[23];
  const float* ff_fc1_w = (const float*)d_in[24];
  const float* ff_fc1_b = (const float*)d_in[25];
  const float* ff_fc2_w = (const float*)d_in[26];
  const float* ff_fc2_b = (const float*)d_in[27];
  const float* out_w    = (const float*)d_in[28];
  const float* out_b    = (const float*)d_in[29];

  // Workspace carve-up (~530 KB total); f16 region is 32B-aligned.
  float* xin  = (float*)d_ws;            // 400*80
  float* zbuf = xin  + LSEQ * D_INNER;   // 400*80
  float* xsb  = zbuf + LSEQ * D_INNER;   // 400*80
  float* dbc  = xsb  + LSEQ * D_INNER;   // 400*35
  float* rmsb = dbc  + LSEQ * 35;        // 400
  _Float16* kfrag = (_Float16*)(rmsb + LSEQ); // 25600 halfs
  _Float16* vfrag = kfrag + KFRAG_N;          // 19968 halfs

  mamba_stack_kernel<<<1, 1024, 0, stream>>>(
      mash, l_norm_w, l_in_w, l_conv_w, l_conv_b, l_xproj, l_dt_w, l_dt_b,
      l_A_log, l_D, l_out_w, ca_lnc_w, ca_lnc_b, ca_kv_w,
      xin, zbuf, xsb, dbc, rmsb, kfrag, vfrag);

  attn_ff_kernel<<<NQ / 32, 64, 0, stream>>>(
      qry, pe_mlp_w, pe_mlp_b, ca_lnq_w, ca_lnq_b, ca_q_w,
      ca_out_w, ca_out_b, ff_ln_w, ff_ln_b,
      ff_fc1_w, ff_fc1_b, ff_fc2_w, ff_fc2_b,
      out_w, out_b, kfrag, vfrag, (float*)d_out);
}
